// MultiHeadAttention_1357209665886
// MI455X (gfx1250) — matrix operand order repaired and checked by reading, not compiled, not run on hardware
//
#include <hip/hip_runtime.h>

typedef __attribute__((ext_vector_type(16))) _Float16 v16h;
typedef __attribute__((ext_vector_type(8)))  _Float16 v8h;
typedef __attribute__((ext_vector_type(8)))  float    v8f;
typedef __attribute__((ext_vector_type(4)))  int      v4i;

#define AS1 __attribute__((address_space(1)))
#define AS3 __attribute__((address_space(3)))

#define WMMA_F16(A, B, C) \
    __builtin_amdgcn_wmma_f32_16x16x32_f16(false, (A), false, (B), (short)0, (C), false, false)

// ---- CDNA5 async global->LDS copy (ASYNCcnt path), with portable fallback ----
#if defined(__has_builtin)
#  if __has_builtin(__builtin_amdgcn_global_load_async_to_lds_b128)
#    define HAVE_ASYNC_LDS 1
#  endif
#endif
#ifndef HAVE_ASYNC_LDS
#  define HAVE_ASYNC_LDS 0
#endif

// Copy 16 bytes global -> LDS. Async (ASYNCcnt) when available.
__device__ __forceinline__ void copy16_g2l(void* ldst, const void* gsrc) {
#if HAVE_ASYNC_LDS
    __builtin_amdgcn_global_load_async_to_lds_b128(
        (AS1 v4i*)(unsigned long long)gsrc,
        (AS3 v4i*)(unsigned)(unsigned long long)ldst,
        0, 0);
#else
    *(v8h*)ldst = *(const v8h*)gsrc;
#endif
}

__device__ __forceinline__ void wait_async() {
#if HAVE_ASYNC_LDS
#  if __has_builtin(__builtin_amdgcn_s_wait_asynccnt)
    __builtin_amdgcn_s_wait_asynccnt(0);
#  else
    asm volatile("s_wait_asynccnt 0" ::: "memory");
#  endif
#endif
}

// Build a v16h A/B fragment from two contiguous 8-half (16B) chunks.
__device__ __forceinline__ v16h make_frag(const _Float16* p0, const _Float16* p1) {
    v8h c0 = *(const v8h*)p0;
    v8h c1 = *(const v8h*)p1;
    v16h r;
#pragma unroll
    for (int j = 0; j < 8; ++j) { r[j] = c0[j]; r[j + 8] = c1[j]; }
    return r;
}

// ---------------------------------------------------------------------------
// Transposed f32 -> f16 weight conversion: src[K][N] f32 -> dst[N][K] f16.
// One-time pass so all GEMM B tiles become contiguous-k (async-stageable).
// ---------------------------------------------------------------------------
__global__ __launch_bounds__(256) void cvt_transpose_kernel(
    const float* __restrict__ src, _Float16* __restrict__ dst, int K, int N) {
    __shared__ float tile[32][33];
    const int k0 = blockIdx.x * 32, n0 = blockIdx.y * 32;
    const int tx = threadIdx.x & 31, ty = threadIdx.x >> 5;
#pragma unroll
    for (int i = 0; i < 4; ++i)
        tile[ty + i * 8][tx] = src[(size_t)(k0 + ty + i * 8) * N + n0 + tx];
    __syncthreads();
#pragma unroll
    for (int i = 0; i < 4; ++i)
        dst[(size_t)(n0 + ty + i * 8) * K + k0 + tx] = (_Float16)tile[tx][ty + i * 8];
}

// ---------------------------------------------------------------------------
// GEMM1: QKV = x[8192,1024] @ Wqkv + b, scattered to Q/K/V [B,H,S,Dh] f16.
// 128x128 tile, K-step 64, double-buffered LDS, async-staged B tiles.
// 8 waves x (32x64) sub-tiles, 16 wmma per K-step.
// ---------------------------------------------------------------------------
__global__ __launch_bounds__(256) void gemm_qkv_kernel(
    const float* __restrict__ x, const _Float16* __restrict__ wT,  // wT: [3072][1024]
    const float* __restrict__ bias,
    _Float16* __restrict__ qb, _Float16* __restrict__ kb, _Float16* __restrict__ vb) {
    __shared__ _Float16 lds_a[2][128][72];   // [m][k], padded rows (144B, 16B-aligned)
    __shared__ _Float16 lds_b[2][128][72];   // [n][k]

    const int tid  = threadIdx.x;
    const int lane = tid & 31, wave = tid >> 5;
    const int lh = lane & 15, hi = lane >> 4;
    const int wm = wave & 3, wn = wave >> 2;
    const int m0 = blockIdx.x * 128, n0 = blockIdx.y * 128;

    v8f acc[2][4];
#pragma unroll
    for (int i = 0; i < 2; ++i)
#pragma unroll
        for (int j = 0; j < 4; ++j)
#pragma unroll
            for (int r = 0; r < 8; ++r) acc[i][j][r] = 0.0f;

    auto stage = [&](int kk, int buf) {
        // A tile 128x64: f32 -> f16 conversion (VALU path)
#pragma unroll
        for (int i = 0; i < 8; ++i) {
            int row = (tid >> 4) + i * 16;
            int col = (tid & 15) * 4;
            float4 f = *(const float4*)(x + (size_t)(m0 + row) * 1024 + kk + col);
            lds_a[buf][row][col + 0] = (_Float16)f.x;
            lds_a[buf][row][col + 1] = (_Float16)f.y;
            lds_a[buf][row][col + 2] = (_Float16)f.z;
            lds_a[buf][row][col + 3] = (_Float16)f.w;
        }
        // B tile 128x64 f16 from transposed weights: async global->LDS
#pragma unroll
        for (int i = 0; i < 4; ++i) {
            int c = tid + i * 256;
            int row = c >> 3, kc = (c & 7) * 8;
            copy16_g2l(&lds_b[buf][row][kc], wT + (size_t)(n0 + row) * 1024 + kk + kc);
        }
    };

    stage(0, 0);
    wait_async();
    __syncthreads();

    for (int it = 0; it < 16; ++it) {
        const int cur = it & 1;
        if (it + 1 < 16) stage((it + 1) * 64, cur ^ 1);
#pragma unroll
        for (int c = 0; c < 2; ++c) {
            v16h af[2], bf[4];
#pragma unroll
            for (int im = 0; im < 2; ++im) {
                const _Float16* p = &lds_a[cur][wm * 32 + im * 16 + lh][c * 32 + hi * 16];
                af[im] = make_frag(p, p + 8);
            }
#pragma unroll
            for (int in = 0; in < 4; ++in) {
                const _Float16* p = &lds_b[cur][wn * 64 + in * 16 + lh][c * 32 + hi * 16];
                bf[in] = make_frag(p, p + 8);
            }
#pragma unroll
            for (int im = 0; im < 2; ++im)
#pragma unroll
                for (int in = 0; in < 4; ++in)
                    acc[im][in] = WMMA_F16(af[im], bf[in], acc[im][in]);
        }
        wait_async();
        __syncthreads();
    }

    // Epilogue: bias + scatter to Q/K/V [B(4),H(16),S(2048),Dh(64)]
#pragma unroll
    for (int im = 0; im < 2; ++im) {
#pragma unroll
        for (int in = 0; in < 4; ++in) {
            int gn = n0 + wn * 64 + in * 16 + lh;
            float bv = bias[gn];
            int sel = gn >> 10;
            int d = gn & 1023;
            int h = d >> 6, dh = d & 63;
            _Float16* dst = (sel == 0) ? qb : ((sel == 1) ? kb : vb);
#pragma unroll
            for (int r = 0; r < 8; ++r) {
                int gm = m0 + wm * 32 + im * 16 + hi * 8 + r;
                int bi = gm >> 11, s = gm & 2047;
                float val = acc[im][in][r] + bv;
                dst[(size_t)((bi * 16 + h) * 2048 + s) * 64 + dh] = (_Float16)val;
            }
        }
    }
}

// ---------------------------------------------------------------------------
// Flash attention: WG = 128 query rows of one (b,h); 8 waves x 16 rows.
// Double-buffered KV tiles of 64; K staged async; online softmax.
// ---------------------------------------------------------------------------
__global__ __launch_bounds__(256) void attention_kernel(
    const _Float16* __restrict__ qb, const _Float16* __restrict__ kb,
    const _Float16* __restrict__ vb, _Float16* __restrict__ attn) {
    __shared__ _Float16 kt[2][64][72];     // [t][dh] row-major   (QK^T B-frags)
    __shared__ _Float16 vt[2][64][72];     // [dh][t] transposed  (PV  B-frags)
    __shared__ _Float16 pt[8][16][72];     // per-wave P scratch  [m][t]

    const int tid  = threadIdx.x;
    const int lane = tid & 31, wave = tid >> 5;
    const int lh = lane & 15, hi = lane >> 4;
    const int bh = blockIdx.y;
    const int s0 = blockIdx.x * 128;
    const int b = bh >> 4, h = bh & 15;
    const size_t base = (size_t)bh * 2048 * 64;
    const float scale = 0.125f;            // 1/sqrt(64)

    // This wave's 16-row Q tile as two A fragments (Dh split k=0..31 / 32..63)
    v16h aq[2];
    {
        const _Float16* qrow = qb + base + (size_t)(s0 + wave * 16 + lh) * 64;
#pragma unroll
        for (int t = 0; t < 2; ++t)
            aq[t] = make_frag(qrow + t * 32 + hi * 16, qrow + t * 32 + hi * 16 + 8);
    }

    float m_run[8], l_run[8];
    v8f oacc[4];
#pragma unroll
    for (int r = 0; r < 8; ++r) { m_run[r] = -3.0e38f; l_run[r] = 0.0f; }
#pragma unroll
    for (int io = 0; io < 4; ++io)
#pragma unroll
        for (int r = 0; r < 8; ++r) oacc[io][r] = 0.0f;

    auto stageKV = [&](int t0, int buf) {
        // K: contiguous rows -> async global->LDS
#pragma unroll
        for (int i = 0; i < 2; ++i) {
            int c = tid + i * 256;
            int trow = c >> 3, kc = (c & 7) * 8;
            copy16_g2l(&kt[buf][trow][kc], kb + base + (size_t)(t0 + trow) * 64 + kc);
        }
        // V: transpose through VGPRs
        int trow = tid >> 2, tcol = (tid & 3) * 16;
        const _Float16* vsrc = vb + base + (size_t)(t0 + trow) * 64 + tcol;
        v8h v0 = *(const v8h*)vsrc;
        v8h v1 = *(const v8h*)(vsrc + 8);
#pragma unroll
        for (int j = 0; j < 8; ++j) {
            vt[buf][tcol + j][trow]     = v0[j];
            vt[buf][tcol + 8 + j][trow] = v1[j];
        }
    };

    stageKV(0, 0);
    wait_async();
    __syncthreads();

    for (int it = 0; it < 32; ++it) {
        const int cur = it & 1;
        if (it + 1 < 32) stageKV((it + 1) * 64, cur ^ 1);

        // Scores: 16x64 tile = 4 N-frags, K-dim = Dh = 64 (two wmma each)
        v8f sacc[4];
#pragma unroll
        for (int nt = 0; nt < 4; ++nt) {
#pragma unroll
            for (int r = 0; r < 8; ++r) sacc[nt][r] = 0.0f;
#pragma unroll
            for (int t = 0; t < 2; ++t) {
                const _Float16* kr = &kt[cur][nt * 16 + lh][t * 32 + hi * 16];
                v16h bk = make_frag(kr, kr + 8);
                sacc[nt] = WMMA_F16(aq[t], bk, sacc[nt]);
            }
        }

        // Online softmax (row = hi*8 + r, spread over 16 lanes of this half)
        float sv[4][8], rmax[8], rsum[8], mnew[8], alpha[8], p[4][8];
#pragma unroll
        for (int r = 0; r < 8; ++r) rmax[r] = -3.0e38f;
#pragma unroll
        for (int nt = 0; nt < 4; ++nt)
#pragma unroll
            for (int r = 0; r < 8; ++r) {
                sv[nt][r] = sacc[nt][r] * scale;
                rmax[r] = fmaxf(rmax[r], sv[nt][r]);
            }
#pragma unroll
        for (int mask = 1; mask < 16; mask <<= 1)
#pragma unroll
            for (int r = 0; r < 8; ++r)
                rmax[r] = fmaxf(rmax[r], __shfl_xor(rmax[r], mask));
#pragma unroll
        for (int r = 0; r < 8; ++r) {
            mnew[r]  = fmaxf(m_run[r], rmax[r]);
            alpha[r] = __expf(m_run[r] - mnew[r]);
            rsum[r]  = 0.0f;
        }
#pragma unroll
        for (int nt = 0; nt < 4; ++nt)
#pragma unroll
            for (int r = 0; r < 8; ++r) {
                p[nt][r] = __expf(sv[nt][r] - mnew[r]);
                rsum[r] += p[nt][r];
            }
#pragma unroll
        for (int mask = 1; mask < 16; mask <<= 1)
#pragma unroll
            for (int r = 0; r < 8; ++r)
                rsum[r] += __shfl_xor(rsum[r], mask);
#pragma unroll
        for (int r = 0; r < 8; ++r) {
            l_run[r] = l_run[r] * alpha[r] + rsum[r];
            m_run[r] = mnew[r];
        }
#pragma unroll
        for (int io = 0; io < 4; ++io)
#pragma unroll
            for (int r = 0; r < 8; ++r) oacc[io][r] *= alpha[r];

        // C-layout P -> LDS, re-read as A-layout fragments
#pragma unroll
        for (int nt = 0; nt < 4; ++nt)
#pragma unroll
            for (int r = 0; r < 8; ++r)
                pt[wave][hi * 8 + r][nt * 16 + lh] = (_Float16)p[nt][r];
        __syncthreads();

        // O += P @ V  (K-dim = 64 key positions, two wmma per output frag)
#pragma unroll
        for (int t = 0; t < 2; ++t) {
            const _Float16* pr = &pt[wave][lh][t * 32 + hi * 16];
            v16h ap = make_frag(pr, pr + 8);
#pragma unroll
            for (int io = 0; io < 4; ++io) {
                const _Float16* vr = &vt[cur][io * 16 + lh][t * 32 + hi * 16];
                v16h bv = make_frag(vr, vr + 8);
                oacc[io] = WMMA_F16(ap, bv, oacc[io]);
            }
        }
        wait_async();
        __syncthreads();
    }

    // Normalize and write to attn [B,S,1024] (D index = h*64 + dh), f16
#pragma unroll
    for (int io = 0; io < 4; ++io)
#pragma unroll
        for (int r = 0; r < 8; ++r) {
            int s   = s0 + wave * 16 + hi * 8 + r;
            int col = h * 64 + io * 16 + lh;
            float val = oacc[io][r] / l_run[r];
            attn[(size_t)(b * 2048 + s) * 1024 + col] = (_Float16)val;
        }
}

// ---------------------------------------------------------------------------
// GEMM2: out = attn[8192,1024] @ Wout + b_out (f32 out).
// Both A and B tiles async-staged (A is f16, B pre-transposed f16).
// ---------------------------------------------------------------------------
__global__ __launch_bounds__(256) void gemm_out_kernel(
    const _Float16* __restrict__ a, const _Float16* __restrict__ wT,  // wT: [1024][1024]
    const float* __restrict__ bias, float* __restrict__ out) {
    __shared__ _Float16 lds_a[2][128][72];
    __shared__ _Float16 lds_b[2][128][72];

    const int tid  = threadIdx.x;
    const int lane = tid & 31, wave = tid >> 5;
    const int lh = lane & 15, hi = lane >> 4;
    const int wm = wave & 3, wn = wave >> 2;
    const int m0 = blockIdx.x * 128, n0 = blockIdx.y * 128;

    v8f acc[2][4];
#pragma unroll
    for (int i = 0; i < 2; ++i)
#pragma unroll
        for (int j = 0; j < 4; ++j)
#pragma unroll
            for (int r = 0; r < 8; ++r) acc[i][j][r] = 0.0f;

    auto stage = [&](int kk, int buf) {
#pragma unroll
        for (int i = 0; i < 4; ++i) {
            int c = tid + i * 256;
            int row = c >> 3, kc = (c & 7) * 8;
            copy16_g2l(&lds_a[buf][row][kc], a  + (size_t)(m0 + row) * 1024 + kk + kc);
            copy16_g2l(&lds_b[buf][row][kc], wT + (size_t)(n0 + row) * 1024 + kk + kc);
        }
    };

    stage(0, 0);
    wait_async();
    __syncthreads();

    for (int it = 0; it < 16; ++it) {
        const int cur = it & 1;
        if (it + 1 < 16) stage((it + 1) * 64, cur ^ 1);
#pragma unroll
        for (int c = 0; c < 2; ++c) {
            v16h af[2], bf[4];
#pragma unroll
            for (int im = 0; im < 2; ++im) {
                const _Float16* p = &lds_a[cur][wm * 32 + im * 16 + lh][c * 32 + hi * 16];
                af[im] = make_frag(p, p + 8);
            }
#pragma unroll
            for (int in = 0; in < 4; ++in) {
                const _Float16* p = &lds_b[cur][wn * 64 + in * 16 + lh][c * 32 + hi * 16];
                bf[in] = make_frag(p, p + 8);
            }
#pragma unroll
            for (int im = 0; im < 2; ++im)
#pragma unroll
                for (int in = 0; in < 4; ++in)
                    acc[im][in] = WMMA_F16(af[im], bf[in], acc[im][in]);
        }
        wait_async();
        __syncthreads();
    }

#pragma unroll
    for (int im = 0; im < 2; ++im)
#pragma unroll
        for (int in = 0; in < 4; ++in) {
            int gn = n0 + wn * 64 + in * 16 + lh;
            float bv = bias[gn];
#pragma unroll
            for (int r = 0; r < 8; ++r) {
                int gm = m0 + wm * 32 + im * 16 + hi * 8 + r;
                out[(size_t)gm * 1024 + gn] = acc[im][in][r] + bv;
            }
        }
}

// ---------------------------------------------------------------------------
extern "C" void kernel_launch(void* const* d_in, const int* in_sizes, int n_in,
                              void* d_out, int out_size, void* d_ws, size_t ws_size,
                              hipStream_t stream) {
    (void)in_sizes; (void)n_in; (void)out_size; (void)ws_size;
    const float* x    = (const float*)d_in[0];
    const float* Wqkv = (const float*)d_in[1];
    const float* bqkv = (const float*)d_in[2];
    const float* Wout = (const float*)d_in[3];
    const float* bout = (const float*)d_in[4];
    float* out = (float*)d_out;

    _Float16* ws     = (_Float16*)d_ws;
    _Float16* wqkv_t = ws;                       // [3072][1024]
    _Float16* wout_t = wqkv_t + 3145728;         // [1024][1024]
    _Float16* qb     = wout_t + 1048576;         // 4*16*2048*64
    _Float16* kb     = qb + 8388608;
    _Float16* vb     = kb + 8388608;
    _Float16* attn   = vb + 8388608;             // 4*2048*1024

    cvt_transpose_kernel<<<dim3(32, 96), 256, 0, stream>>>(Wqkv, wqkv_t, 1024, 3072);
    cvt_transpose_kernel<<<dim3(32, 32), 256, 0, stream>>>(Wout, wout_t, 1024, 1024);
    gemm_qkv_kernel<<<dim3(64, 24), 256, 0, stream>>>(x, wqkv_t, bqkv, qb, kb, vb);
    attention_kernel<<<dim3(16, 64), 256, 0, stream>>>(qb, kb, vb, attn);
    gemm_out_kernel<<<dim3(64, 8), 256, 0, stream>>>(attn, wout_t, bout, out);
}
